// RoPEAttention_27728308863269
// MI455X (gfx1250) — matrix operand order repaired and checked by reading, not compiled, not run on hardware
//
#include <hip/hip_runtime.h>
#include <hip/hip_bf16.h>

typedef _Float16 f16;
typedef __attribute__((ext_vector_type(16))) _Float16 v16h;
typedef __attribute__((ext_vector_type(8)))  _Float16 v8h;
typedef __attribute__((ext_vector_type(4)))  _Float16 v4h;
typedef __attribute__((ext_vector_type(8)))  float    v8f;
typedef __attribute__((ext_vector_type(4)))  int      v4i;

#define BB 4
#define NN 4096
#define CC 256
#define HH 4
#define DD 64

#define AS1 __attribute__((address_space(1)))
#define AS3 __attribute__((address_space(3)))

// CDNA5 async global->LDS path (ASYNCcnt-tracked), guarded so both compiler
// passes and older toolchains still build (falls back to sync LDS staging).
#if __has_builtin(__builtin_amdgcn_global_load_async_to_lds_b128) && \
    __has_builtin(__builtin_amdgcn_s_wait_asynccnt)
#define USE_ASYNC_LDS 1
#else
#define USE_ASYNC_LDS 0
#endif

static __device__ __forceinline__ void stage_b128(const f16* g, f16* l) {
#if USE_ASYNC_LDS
  __builtin_amdgcn_global_load_async_to_lds_b128((AS1 v4i*)g, (AS3 v4i*)l, 0,
                                                 0);
#else
  *(v8h*)l = *(const v8h*)g;
#endif
}

static __device__ __forceinline__ void wait_async_le4() {
#if USE_ASYNC_LDS
  __builtin_amdgcn_s_wait_asynccnt(4);
#endif
}
static __device__ __forceinline__ void wait_async_0() {
#if USE_ASYNC_LDS
  __builtin_amdgcn_s_wait_asynccnt(0);
#endif
}

static __device__ __forceinline__ v8f zero8() {
  v8f z;
#pragma unroll
  for (int i = 0; i < 8; ++i) z[i] = 0.0f;
  return z;
}

static __device__ __forceinline__ v16h cat16(v8h lo, v8h hi) {
  v16h r;
#pragma unroll
  for (int i = 0; i < 8; ++i) { r[i] = lo[i]; r[i + 8] = hi[i]; }
  return r;
}

// Load 2x8 consecutive f32 and convert to a v16h WMMA operand fragment.
static __device__ __forceinline__ v16h cvt16(const float* __restrict__ p0,
                                             const float* __restrict__ p1) {
  float4 a0 = *(const float4*)(p0);
  float4 a1 = *(const float4*)(p0 + 4);
  float4 a2 = *(const float4*)(p1);
  float4 a3 = *(const float4*)(p1 + 4);
  v16h r;
  r[0]  = (f16)a0.x; r[1]  = (f16)a0.y; r[2]  = (f16)a0.z; r[3]  = (f16)a0.w;
  r[4]  = (f16)a1.x; r[5]  = (f16)a1.y; r[6]  = (f16)a1.z; r[7]  = (f16)a1.w;
  r[8]  = (f16)a2.x; r[9]  = (f16)a2.y; r[10] = (f16)a2.z; r[11] = (f16)a2.w;
  r[12] = (f16)a3.x; r[13] = (f16)a3.y; r[14] = (f16)a3.z; r[15] = (f16)a3.w;
  return r;
}

// ---------------------------------------------------------------------------
// Kernel 0: one-shot f32 -> f16 conversion of the four weight matrices.
// ---------------------------------------------------------------------------
__global__ __launch_bounds__(256) void cvt_w_kernel(
    const float* __restrict__ Wq, const float* __restrict__ Wk,
    const float* __restrict__ Wv, const float* __restrict__ Wo,
    f16* __restrict__ Wqh, f16* __restrict__ Wkh, f16* __restrict__ Wvh,
    f16* __restrict__ Woh) {
  const int sel = blockIdx.y;
  const float* src = (sel == 0) ? Wq : (sel == 1) ? Wk : (sel == 2) ? Wv : Wo;
  f16* dst = (sel == 0) ? Wqh : (sel == 1) ? Wkh : (sel == 2) ? Wvh : Woh;
  const int i = (blockIdx.x * 256 + threadIdx.x) * 4;
  float4 x = *(const float4*)(src + i);
  v4h o;
  o[0] = (f16)x.x; o[1] = (f16)x.y; o[2] = (f16)x.z; o[3] = (f16)x.w;
  *(v4h*)(dst + i) = o;
}

// ---------------------------------------------------------------------------
// Kernel 1: fused QKV projection (WMMA f16) + bias + axial RoPE.
//   grid.x = (B*N)/64 row tiles, grid.y = 3 (q,k,v). 256 threads = 8 waves.
// Outputs: Qh,Kh [B,H,N,64] f16 (roped); Vt [B,H,64,N] f16 (transposed).
// ---------------------------------------------------------------------------
__global__ __launch_bounds__(256) void qkv_proj_rope_kernel(
    const float* __restrict__ q, const float* __restrict__ k,
    const float* __restrict__ v,
    const f16* __restrict__ Wqh, const float* __restrict__ bq,
    const f16* __restrict__ Wkh, const float* __restrict__ bk,
    const f16* __restrict__ Wvh, const float* __restrict__ bv,
    const int* __restrict__ nk_excl_p,
    f16* __restrict__ Qh, f16* __restrict__ Kh, f16* __restrict__ Vt) {
  const int proj = blockIdx.y;
  const float* X  = (proj == 0) ? q   : (proj == 1) ? k   : v;
  const f16* W    = (proj == 0) ? Wqh : (proj == 1) ? Wkh : Wvh;
  const float* bias = (proj == 0) ? bq : (proj == 1) ? bk : bv;

  const int tid  = threadIdx.x;
  const int lane = tid & 31;
  const int wave = tid >> 5;
  const int l16  = lane & 15;
  const int hi   = lane >> 4;
  const int row0 = blockIdx.x * 64 + (wave & 3) * 16;
  const int col0 = (wave >> 2) * 128;

  v8f acc[8];
#pragma unroll
  for (int t = 0; t < 8; ++t) acc[t] = zero8();

  const float* xrow = X + (size_t)(row0 + l16) * CC;
  for (int c = 0; c < CC; c += 32) {
    v16h A = cvt16(xrow + c + hi * 8, xrow + c + 16 + hi * 8);
#pragma unroll
    for (int t = 0; t < 8; ++t) {
      const int ocol = col0 + t * 16 + l16;
      const v8h* wp = (const v8h*)(W + (size_t)ocol * CC + c + hi * 8);
      v16h Bv = cat16(wp[0], wp[2]);
      acc[t] = __builtin_amdgcn_wmma_f32_16x16x32_f16(
          false, A, false, Bv, (short)0, acc[t], false, false);
    }
  }

  const int nkr = NN - nk_excl_p[0];
  const float LOG_TH_O16 = 9.2103403719761836f / 16.0f;  // ln(10000)/16

#pragma unroll
  for (int t = 0; t < 8; ++t) {
    const int ocol = col0 + t * 16 + l16;
    const int h = ocol >> 6;
    const int d = ocol & 63;
    const float bval = bias[ocol];
    const int jp = d >> 1;  // rotary pair index 0..31
    const float fr = __expf(-(float)((jp < 16) ? jp : (jp - 16)) * LOG_TH_O16);
#pragma unroll
    for (int r = 0; r < 8; ++r) {
      const int row = row0 + r + hi * 8;  // C layout: VGPR r -> M=r / M=8+r
      const int b = row >> 12;
      const int n = row & (NN - 1);
      float y = acc[t][r] + bval;
      float part = __shfl_xor(y, 1, 32);  // rotary partner (d^1) in lane^1
      if (proj == 2) {
        Vt[(((size_t)b * HH + h) * DD + d) * NN + n] = (f16)y;
      } else {
        float out = y;
        if (proj == 0 || n < nkr) {
          const float pos = (jp < 16) ? (float)(n & 63) : (float)(n >> 6);
          float sn, cs;
          __sincosf(pos * fr, &sn, &cs);
          out = (d & 1) ? (part * sn + y * cs) : (y * cs - part * sn);
        }
        f16* dst = (proj == 0) ? Qh : Kh;
        dst[(((size_t)b * HH + h) * NN + n) * DD + d] = (f16)out;
      }
    }
  }
}

// ---------------------------------------------------------------------------
// Kernel 2: flash attention. One (b,h) per blockIdx.x, 128 query rows per
// blockIdx.y; each wave owns 16 rows (no cross-wave softmax reductions).
// K/V tiles (64x64 f16 each) staged into LDS once per block via the CDNA5
// async global->LDS engine, double-buffered: issue tile kt+1, wait
// asynccnt<=4 (tile kt landed), barrier, compute, barrier.
// ---------------------------------------------------------------------------
__global__ __launch_bounds__(256) void flash_attn_kernel(
    const f16* __restrict__ Qh, const f16* __restrict__ Kh,
    const f16* __restrict__ Vt, f16* __restrict__ Of) {
  const int bh = blockIdx.x;  // b*H + h
  const int tid = threadIdx.x;
  const int lane = tid & 31, wave = tid >> 5;
  const int l16 = lane & 15, hi = lane >> 4;
  const int q0 = blockIdx.y * 128 + wave * 16;

  const f16* Qb = Qh + (size_t)bh * NN * DD;
  const f16* Kb = Kh + (size_t)bh * NN * DD;
  const f16* Vb = Vt + (size_t)bh * DD * NN;  // [d][n]

  __shared__ __align__(16) f16 Ktile[2][64][64];  // [key][d]
  __shared__ __align__(16) f16 Vtile[2][64][64];  // [d][key]
  __shared__ __align__(16) f16 Plds[8][16][64];   // wave-private P

  // Cooperative staging: 256 threads x (2 chunks K + 2 chunks V) of 16B.
  const int c0 = tid * 2;           // chunks 2t, 2t+1 of 512 (same row)
  const int srow = c0 >> 3;         // 0..63
  const int scol = (c0 & 7) * 8;    // 0,16,32,48

  // Preload this wave's Q fragment; fold in softmax scale 1/8 (exact in f16).
  const f16* qr = Qb + (size_t)(q0 + l16) * DD;
  v16h Aq[2];
#pragma unroll
  for (int ki = 0; ki < 2; ++ki) {
    v8h a0 = *(const v8h*)(qr + ki * 32 + hi * 8);
    v8h a1 = *(const v8h*)(qr + ki * 32 + 16 + hi * 8);
    v16h a = cat16(a0, a1);
#pragma unroll
    for (int i = 0; i < 16; ++i) a[i] = a[i] * (f16)0.125f;
    Aq[ki] = a;
  }

  v8f Oacc[4];
  float m[8], l[8];
#pragma unroll
  for (int t = 0; t < 4; ++t) Oacc[t] = zero8();
#pragma unroll
  for (int r = 0; r < 8; ++r) { m[r] = -3.0e38f; l[r] = 0.0f; }

  auto stage = [&](int kt, int buf) {
    const int k0 = kt * 64;
    const f16* gk = Kb + (size_t)(k0 + srow) * DD + scol;
    const f16* gv = Vb + (size_t)srow * NN + k0 + scol;
    stage_b128(gk,     &Ktile[buf][srow][scol]);
    stage_b128(gk + 8, &Ktile[buf][srow][scol + 8]);
    stage_b128(gv,     &Vtile[buf][srow][scol]);
    stage_b128(gv + 8, &Vtile[buf][srow][scol + 8]);
  };

  stage(0, 0);

  for (int kt = 0; kt < NN / 64; ++kt) {
    const int buf = kt & 1;
    if (kt + 1 < NN / 64) {
      stage(kt + 1, buf ^ 1);  // prefetch next tile into the other buffer
      wait_async_le4();        // tile kt's 4 async ops have landed (in-order)
    } else {
      wait_async_0();
    }
    __syncthreads();  // all waves' staged data visible

    v8f S[4];
#pragma unroll
    for (int j2 = 0; j2 < 4; ++j2) S[j2] = zero8();

    // S = (Q*scale) @ K^T : B-operand lane n = LDS K row along d.
#pragma unroll
    for (int j2 = 0; j2 < 4; ++j2) {
#pragma unroll
      for (int ki = 0; ki < 2; ++ki) {
        const v8h* kp =
            (const v8h*)&Ktile[buf][j2 * 16 + l16][ki * 32 + hi * 8];
        v16h Bv = cat16(kp[0], kp[2]);
        S[j2] = __builtin_amdgcn_wmma_f32_16x16x32_f16(
            false, Aq[ki], false, Bv, (short)0, S[j2], false, false);
      }
    }

    // Online softmax: row stats via half-wave shfl reductions.
    float mt[8];
#pragma unroll
    for (int r = 0; r < 8; ++r)
      mt[r] = fmaxf(fmaxf(S[0][r], S[1][r]), fmaxf(S[2][r], S[3][r]));
#pragma unroll
    for (int msk = 1; msk <= 8; msk <<= 1)
#pragma unroll
      for (int r = 0; r < 8; ++r)
        mt[r] = fmaxf(mt[r], __shfl_xor(mt[r], msk, 32));

    float scl[8], rs[8];
#pragma unroll
    for (int r = 0; r < 8; ++r) {
      const float mn = fmaxf(m[r], mt[r]);
      scl[r] = __expf(m[r] - mn);
      m[r] = mn;
      rs[r] = 0.0f;
    }

#pragma unroll
    for (int j2 = 0; j2 < 4; ++j2)
#pragma unroll
      for (int r = 0; r < 8; ++r) {
        const float p = __expf(S[j2][r] - m[r]);
        rs[r] += p;
        Plds[wave][r + hi * 8][j2 * 16 + l16] = (f16)p;
      }
#pragma unroll
    for (int msk = 1; msk <= 8; msk <<= 1)
#pragma unroll
      for (int r = 0; r < 8; ++r) rs[r] += __shfl_xor(rs[r], msk, 32);
#pragma unroll
    for (int r = 0; r < 8; ++r) l[r] = l[r] * scl[r] + rs[r];

#pragma unroll
    for (int t = 0; t < 4; ++t)
#pragma unroll
      for (int r = 0; r < 8; ++r) Oacc[t][r] *= scl[r];

    // O += P @ V : A from wave-private LDS P (same-wave DS ops are in-order,
    // ISA 7.3), B lane n = LDS V row (d) along keys.
#pragma unroll
    for (int ki = 0; ki < 2; ++ki) {
      v8h p0 = *(const v8h*)&Plds[wave][l16][ki * 32 + hi * 8];
      v8h p1 = *(const v8h*)&Plds[wave][l16][ki * 32 + 16 + hi * 8];
      v16h Ap = cat16(p0, p1);
#pragma unroll
      for (int j2 = 0; j2 < 4; ++j2) {
        const v8h* vp =
            (const v8h*)&Vtile[buf][j2 * 16 + l16][ki * 32 + hi * 8];
        v16h Bv = cat16(vp[0], vp[2]);
        Oacc[j2] = __builtin_amdgcn_wmma_f32_16x16x32_f16(
            false, Ap, false, Bv, (short)0, Oacc[j2], false, false);
      }
    }

    __syncthreads();  // everyone done reading `buf` before it is re-staged
  }

  // Normalize and write O as f16 rows [B,N,256] (head-concat layout).
  const int b = bh >> 2, h = bh & 3;
#pragma unroll
  for (int r = 0; r < 8; ++r) {
    const float inv = 1.0f / l[r];
    const int n = q0 + r + hi * 8;
#pragma unroll
    for (int t = 0; t < 4; ++t) {
      Of[((size_t)b * NN + n) * CC + h * DD + t * 16 + l16] =
          (f16)(Oacc[t][r] * inv);
    }
  }
}

// ---------------------------------------------------------------------------
// Kernel 3: output projection  out = Of @ Wo^T + bo  (f32 result).
// ---------------------------------------------------------------------------
__global__ __launch_bounds__(256) void out_proj_kernel(
    const f16* __restrict__ Of, const f16* __restrict__ Woh,
    const float* __restrict__ bo, float* __restrict__ out) {
  const int tid = threadIdx.x;
  const int lane = tid & 31, wave = tid >> 5;
  const int l16 = lane & 15, hi = lane >> 4;
  const int row0 = blockIdx.x * 64 + (wave & 3) * 16;
  const int col0 = (wave >> 2) * 128;

  v8f acc[8];
#pragma unroll
  for (int t = 0; t < 8; ++t) acc[t] = zero8();

  const f16* orow = Of + (size_t)(row0 + l16) * CC;
  for (int c = 0; c < CC; c += 32) {
    v8h a0 = *(const v8h*)(orow + c + hi * 8);
    v8h a1 = *(const v8h*)(orow + c + 16 + hi * 8);
    v16h A = cat16(a0, a1);
#pragma unroll
    for (int t = 0; t < 8; ++t) {
      const int ocol = col0 + t * 16 + l16;
      const v8h* wp = (const v8h*)(Woh + (size_t)ocol * CC + c + hi * 8);
      v16h Bv = cat16(wp[0], wp[2]);
      acc[t] = __builtin_amdgcn_wmma_f32_16x16x32_f16(
          false, A, false, Bv, (short)0, acc[t], false, false);
    }
  }

#pragma unroll
  for (int t = 0; t < 8; ++t) {
    const int ocol = col0 + t * 16 + l16;
    const float bval = bo[ocol];
#pragma unroll
    for (int r = 0; r < 8; ++r) {
      const int row = row0 + r + hi * 8;
      out[(size_t)row * CC + ocol] = acc[t][r] + bval;
    }
  }
}

// ---------------------------------------------------------------------------
extern "C" void kernel_launch(void* const* d_in, const int* in_sizes, int n_in,
                              void* d_out, int out_size, void* d_ws,
                              size_t ws_size, hipStream_t stream) {
  const float* q  = (const float*)d_in[0];
  const float* k  = (const float*)d_in[1];
  const float* v  = (const float*)d_in[2];
  const float* Wq = (const float*)d_in[3];
  const float* bq = (const float*)d_in[4];
  const float* Wk = (const float*)d_in[5];
  const float* bk = (const float*)d_in[6];
  const float* Wv = (const float*)d_in[7];
  const float* bv = (const float*)d_in[8];
  const float* Wo = (const float*)d_in[9];
  const float* bo = (const float*)d_in[10];
  const int*   nk = (const int*)d_in[11];
  float* out = (float*)d_out;

  const size_t per = (size_t)BB * HH * NN * DD;  // 4,194,304 f16 = 8 MB
  f16* Qh  = (f16*)d_ws;
  f16* Kh  = Qh + per;
  f16* Vt  = Kh + per;
  f16* Of  = Vt + per;
  f16* Wqh = Of + per;          // 4 x 65536 f16 = 512 KB
  f16* Wkh = Wqh + CC * CC;
  f16* Wvh = Wkh + CC * CC;
  f16* Woh = Wvh + CC * CC;

  cvt_w_kernel<<<dim3(64, 4), 256, 0, stream>>>(Wq, Wk, Wv, Wo, Wqh, Wkh, Wvh,
                                                Woh);
  qkv_proj_rope_kernel<<<dim3((BB * NN) / 64, 3), 256, 0, stream>>>(
      q, k, v, Wqh, bq, Wkh, bk, Wvh, bv, nk, Qh, Kh, Vt);
  flash_attn_kernel<<<dim3(BB * HH, NN / 128), 256, 0, stream>>>(Qh, Kh, Vt,
                                                                 Of);
  out_proj_kernel<<<dim3((BB * NN) / 64), 256, 0, stream>>>(Of, Woh, bo, out);
}
